// VectorizedQuantumFluxGNN_50122268344537
// MI455X (gfx1250) — compile-verified, matrix operand order checked by reading, not run on hardware
//
#include <hip/hip_runtime.h>
#include <hip/hip_bf16.h>
#include <math.h>

typedef __attribute__((ext_vector_type(16))) _Float16 v16h;
typedef __attribute__((ext_vector_type(8)))  _Float16 v8h;
typedef __attribute__((ext_vector_type(8)))  float    v8f;

constexpr int   Bn = 16, Sn = 512, Dn = 64, Vn = 32000;
constexpr float EPSF = 1e-12f;

// ---------------- workspace layout (bytes) ----------------
constexpr size_t EMB16_OFF = 0;                                   // [B,S,D] f16  (1 MB)
constexpr size_t N2_OFF    = EMB16_OFF + (size_t)Bn*Sn*Dn*2;      // [B,S]  f32   (32 KB)
constexpr size_t ATTN_OFF  = N2_OFF    + (size_t)Bn*Sn*4;         // [B,S,S] f16  (8 MB)
constexpr size_t GT_OFF    = ATTN_OFF  + (size_t)Bn*Sn*Sn*2;      // [B,D,S] f16  (1 MB, transposed)
constexpr size_t H1_OFF    = GT_OFF    + (size_t)Bn*Sn*Dn*2;      // [B,S,D] f16
constexpr size_t H2_OFF    = H1_OFF    + (size_t)Bn*Sn*Dn*2;      // [B,S,D] f16
constexpr size_t OUTW_OFF  = H2_OFF    + (size_t)Bn*Sn*Dn*2;      // [V,D]  f16   (4 MB)

// ---------------- WMMA helpers ----------------
__device__ inline v8f wmma_f16(v16h a, v16h b, v8f c) {
  return __builtin_amdgcn_wmma_f32_16x16x32_f16(false, a, false, b, (short)0, c,
                                                false, false);
}

// A-fragment (16xK tile, row-major source, lane%16 = row M) or
// B-fragment where column n of B equals row n of a row-major source.
// CDNA5 16-bit layout: lanes 0-15 hold K=0..7 & 16..23, lanes 16-31 hold K=8..15 & 24..31.
__device__ inline v16h frag_row_f16(const _Float16* base, int ld, int lane, int kofs) {
  int r  = lane & 15;
  int kb = (lane & 16) ? 8 : 0;
  const _Float16* p = base + (size_t)r * ld + kofs + kb;
  v16h f;
#pragma unroll
  for (int j = 0; j < 8; ++j) { f[j] = p[j]; f[8 + j] = p[16 + j]; }
  return f;
}

// Same as above but converting from an fp32 row-major source (weights).
__device__ inline v16h frag_row_f32(const float* base, int ld, int lane, int kofs) {
  int r  = lane & 15;
  int kb = (lane & 16) ? 8 : 0;
  const float* p = base + (size_t)r * ld + kofs + kb;
  v16h f;
#pragma unroll
  for (int j = 0; j < 8; ++j) { f[j] = (_Float16)p[j]; f[8 + j] = (_Float16)p[16 + j]; }
  return f;
}

__device__ inline float wave_sum(float v) {
#pragma unroll
  for (int m = 16; m >= 1; m >>= 1) v += __shfl_xor(v, m, 32);
  return v;
}
__device__ inline float wave_max(float v) {
#pragma unroll
  for (int m = 16; m >= 1; m >>= 1) v = fmaxf(v, __shfl_xor(v, m, 32));
  return v;
}

// ---------------- 1. embed + L2 normalize ----------------
// one wave32 per (b,s) row; lane l owns dims 2l, 2l+1
__global__ void k_embed(const int* __restrict__ tokens,
                        const float* __restrict__ temb,
                        _Float16* __restrict__ emb16,
                        float* __restrict__ n2) {
  int row  = blockIdx.x * (blockDim.x >> 5) + (threadIdx.x >> 5);
  int lane = threadIdx.x & 31;
  if (row >= Bn * Sn) return;
  int s   = row & (Sn - 1);
  int tok = tokens[row];
  float v0, v1;
  if (lane == 0) {
    float th = 6.2831853071795864769f * ((float)s / (float)Sn);
    float rs = 0.3f + 0.6f * ((float)s / (float)(Sn - 1));
    v0 = rs * cosf(th);
    v1 = rs * sinf(th);
  } else {
    const float* tr = temb + (size_t)tok * Dn;   // uses dims 0..61 of the row
    v0 = tr[2 * lane - 2];
    v1 = tr[2 * lane - 1];
  }
  float ss   = wave_sum(v0 * v0 + v1 * v1);
  float nrm2 = fmaxf(ss, EPSF);
  float inv  = rsqrtf(nrm2);
  _Float16* er = emb16 + (size_t)row * Dn;
  er[2 * lane]     = (_Float16)(v0 * inv);
  er[2 * lane + 1] = (_Float16)(v1 * inv);
  if (lane == 0) n2[row] = ss / nrm2;           // == |emb|^2 after normalize (~1)
}

// ---------------- 2. distance attention + softmax ----------------
// block = 128 threads (4 waves) per (b, q-tile of 16 rows)
__global__ void k_attention(const _Float16* __restrict__ emb16,
                            const float* __restrict__ n2,
                            _Float16* __restrict__ attn16) {
  __shared__ float nd[16][Sn];                  // 32 KB: -dist tile
  int b     = blockIdx.x >> 5;
  int qt    = blockIdx.x & 31;
  int qbase = qt * 16;
  int lane  = threadIdx.x & 31;
  int wave  = threadIdx.x >> 5;                 // 0..3
  const _Float16* embB = emb16 + (size_t)b * Sn * Dn;
  const float*    n2B  = n2 + (size_t)b * Sn;

  v16h a0 = frag_row_f16(embB + (size_t)qbase * Dn, Dn, lane, 0);
  v16h a1 = frag_row_f16(embB + (size_t)qbase * Dn, Dn, lane, 32);

  for (int kt = wave; kt < Sn / 16; kt += 4) {
    int kbase = kt * 16;
    v16h b0 = frag_row_f16(embB + (size_t)kbase * Dn, Dn, lane, 0);
    v16h b1 = frag_row_f16(embB + (size_t)kbase * Dn, Dn, lane, 32);
    v8f c = {};
    c = wmma_f16(a0, b0, c);
    c = wmma_f16(a1, b1, c);
    int   n   = lane & 15;
    int   mo  = (lane & 16) ? 8 : 0;
    float n2k = n2B[kbase + n];
#pragma unroll
    for (int r = 0; r < 8; ++r) {
      int   m    = r + mo;
      float sq   = n2B[qbase + m] + n2k - 2.0f * c[r];
      float dist = __builtin_amdgcn_sqrtf(fmaxf(sq, EPSF));  // raw v_sqrt_f32
      if (qbase + m == kbase + n) dist = 0.0f;  // eye mask
      nd[m][kbase + n] = -dist;                 // TEMP = 1
    }
  }
  __syncthreads();

  // softmax: each wave owns 4 rows
  for (int rr = 0; rr < 4; ++rr) {
    int   m  = wave * 4 + rr;
    float mx = -1e30f;
    for (int k = lane; k < Sn; k += 32) mx = fmaxf(mx, nd[m][k]);
    mx = wave_max(mx);
    float sm = 0.0f;
    for (int k = lane; k < Sn; k += 32) sm += __expf(nd[m][k] - mx);
    sm = wave_sum(sm);
    float inv = 1.0f / sm;
    _Float16* out = attn16 + ((size_t)b * Sn + qbase + m) * Sn;
    for (int k = lane; k < Sn; k += 32)
      out[k] = (_Float16)(__expf(nd[m][k] - mx) * inv);
  }
}

// ---------------- 3a. g^T = (h @ W^T)^T ----------------
// [B*S,64] x [64,64], stored transposed per batch: gt[b][n][s].
// With the transposed layout each lane's 8 output rows are 8 consecutive s
// positions -> one aligned 16-byte store per lane.
__global__ void k_hw(const _Float16* __restrict__ h,
                     const float* __restrict__ w,
                     _Float16* __restrict__ gt) {
  int idx  = blockIdx.x * (blockDim.x >> 5) + (threadIdx.x >> 5);
  int lane = threadIdx.x & 31;
  int mt = idx >> 2, nt = idx & 3;              // 512 m-tiles x 4 n-tiles
  const _Float16* A = h + (size_t)mt * 16 * Dn;
  const float*    W = w + (size_t)nt * 16 * Dn; // B col n = W row n (W^T)
  v8f c = {};
  c = wmma_f16(frag_row_f16(A, Dn, lane, 0),  frag_row_f32(W, Dn, lane, 0),  c);
  c = wmma_f16(frag_row_f16(A, Dn, lane, 32), frag_row_f32(W, Dn, lane, 32), c);
  int n = lane & 15, mo = (lane & 16) ? 8 : 0;
  int bb    = mt >> 5;                          // 32 m-tiles per batch
  int sbase = (mt * 16) & (Sn - 1);
  v8h o;
#pragma unroll
  for (int r = 0; r < 8; ++r) o[r] = (_Float16)c[r];
  *(v8h*)(gt + ((size_t)bb * Dn + nt * 16 + n) * Sn + sbase + mo) = o;
}

// ---------------- 3b. h' = scale * attn @ g  (per batch, K=512) ----------------
// B-fragment for column n of g == row n of gt -> contiguous row load.
__global__ void k_attnmul(const _Float16* __restrict__ attn16,
                          const _Float16* __restrict__ gt,
                          const float* __restrict__ escale,
                          _Float16* __restrict__ hout) {
  int idx  = blockIdx.x * (blockDim.x >> 5) + (threadIdx.x >> 5);
  int lane = threadIdx.x & 31;
  int b = idx >> 7, rem = idx & 127;
  int mt = rem >> 2, nt = rem & 3;
  const _Float16* Arow = attn16 + ((size_t)b * Sn + mt * 16) * Sn;
  const _Float16* Gt   = gt + ((size_t)b * Dn + nt * 16) * Sn;
  v8f c = {};
  for (int kt = 0; kt < Sn / 32; ++kt) {        // 16 WMMA k-steps
    v16h a  = frag_row_f16(Arow, Sn, lane, kt * 32);
    v16h bm = frag_row_f16(Gt,   Sn, lane, kt * 32);
    c = wmma_f16(a, bm, c);
  }
  float sc = escale[0];
  int n = lane & 15, mo = (lane & 16) ? 8 : 0;
#pragma unroll
  for (int r = 0; r < 8; ++r)
    hout[((size_t)b * Sn + mt * 16 + r + mo) * Dn + nt * 16 + n] =
        (_Float16)(sc * c[r]);
}

// ---------------- 4. out_w fp32 -> f16 ----------------
__global__ void k_cvt(const float* __restrict__ src, _Float16* __restrict__ dst, int n) {
  int i = blockIdx.x * blockDim.x + threadIdx.x;
  if (i < n) dst[i] = (_Float16)src[i];
}

// ---------------- 5. logits = h @ out_w^T  (M=8192, N=32000, K=64) ----------------
// one wave per (m-tile, group of 8 v-tiles); A fragments reused 8x.
// 1.05 GB streaming fp32 output -> nontemporal stores (keep out_w/h in cache).
__global__ void k_logits(const _Float16* __restrict__ h,
                         const _Float16* __restrict__ outw,
                         float* __restrict__ out) {
  int idx  = blockIdx.x * (blockDim.x >> 5) + (threadIdx.x >> 5);
  int lane = threadIdx.x & 31;
  int mt = idx / 250;                           // 512 m-tiles
  int ng = idx % 250;                           // 250 groups of 8 v-tiles
  const _Float16* A = h + (size_t)mt * 16 * Dn;
  v16h a0 = frag_row_f16(A, Dn, lane, 0);
  v16h a1 = frag_row_f16(A, Dn, lane, 32);
  int n = lane & 15, mo = (lane & 16) ? 8 : 0;
#pragma unroll
  for (int t = 0; t < 8; ++t) {
    int vbase = (ng * 8 + t) * 16;
    const _Float16* Wt = outw + (size_t)vbase * Dn;
    if (t < 7) __builtin_prefetch(Wt + 16 * Dn, 0, 0);   // global_prefetch_b8
    v8f c = {};
    c = wmma_f16(a0, frag_row_f16(Wt, Dn, lane, 0),  c);
    c = wmma_f16(a1, frag_row_f16(Wt, Dn, lane, 32), c);
#pragma unroll
    for (int r = 0; r < 8; ++r)
      __builtin_nontemporal_store(
          c[r], &out[(size_t)(mt * 16 + r + mo) * Vn + vbase + n]);
  }
}

// ---------------- launch ----------------
extern "C" void kernel_launch(void* const* d_in, const int* in_sizes, int n_in,
                              void* d_out, int out_size, void* d_ws, size_t ws_size,
                              hipStream_t stream) {
  const int*   tokens   = (const int*)  d_in[0];
  const float* temb     = (const float*)d_in[1];
  const float* lin_w1   = (const float*)d_in[2];
  const float* lin_w2   = (const float*)d_in[3];
  const float* escale   = (const float*)d_in[4];
  const float* out_w    = (const float*)d_in[5];
  float*       logits   = (float*)d_out;

  char* ws = (char*)d_ws;
  _Float16* emb16  = (_Float16*)(ws + EMB16_OFF);
  float*    n2     = (float*)   (ws + N2_OFF);
  _Float16* attn16 = (_Float16*)(ws + ATTN_OFF);
  _Float16* gt16   = (_Float16*)(ws + GT_OFF);
  _Float16* h1     = (_Float16*)(ws + H1_OFF);
  _Float16* h2     = (_Float16*)(ws + H2_OFF);
  _Float16* outw16 = (_Float16*)(ws + OUTW_OFF);

  // 1) embed+normalize: 8192 rows, 8 waves/block
  k_embed<<<(Bn * Sn) / 8, 256, 0, stream>>>(tokens, temb, emb16, n2);

  // 2) attention: one block per (b, q-tile)
  k_attention<<<Bn * (Sn / 16), 128, 0, stream>>>(emb16, n2, attn16);

  // 3) layer 1
  k_hw<<<2048 / 8, 256, 0, stream>>>(emb16, lin_w1, gt16);
  k_attnmul<<<2048 / 8, 256, 0, stream>>>(attn16, gt16, escale, h1);
  //    layer 2
  k_hw<<<2048 / 8, 256, 0, stream>>>(h1, lin_w2, gt16);
  k_attnmul<<<2048 / 8, 256, 0, stream>>>(attn16, gt16, escale, h2);

  // 4) weight conversion
  k_cvt<<<(Vn * Dn + 255) / 256, 256, 0, stream>>>(out_w, outw16, Vn * Dn);

  // 5) logits: 512 * 250 waves, 8 waves/block
  k_logits<<<(512 * 250) / 8, 256, 0, stream>>>(h2, outw16, logits);
}
